// WaveHybridAttention_29308856827977
// MI455X (gfx1250) — matrix operand order repaired and checked by reading, not compiled, not run on hardware
//
#include <hip/hip_runtime.h>

typedef __bf16 bf16;
typedef bf16 v16bf __attribute__((ext_vector_type(16)));
typedef bf16 v8bf  __attribute__((ext_vector_type(8)));
typedef float v8f  __attribute__((ext_vector_type(8)));

#define D_MODEL 1024
#define SEQ     1024
#define BATCH   4
#define N_TOTAL 16
#define N_WAVE  4
#define N_STD   12
#define DK      64
#define STD_DIM 768
#define WAVE_DIM 24
#define WAVE_CHUNK 6

// ---------------------------------------------------------------------------
// WMMA helpers: D = A(16x32) * B(32x16) + C, bf16 inputs, f32 accumulate.
// ---------------------------------------------------------------------------
__device__ __forceinline__ v8f wmma_bf16(v16bf a, v16bf b, v8f c) {
    return __builtin_amdgcn_wmma_f32_16x16x32_bf16(
        /*neg_a=*/false, a, /*neg_b=*/false, b,
        /*c_mod=*/(short)0, c, /*reuse_a=*/false, /*reuse_b=*/false);
}

__device__ __forceinline__ v8f zero8() {
    v8f z;
#pragma unroll
    for (int i = 0; i < 8; ++i) z[i] = 0.0f;
    return z;
}

// A operand (16x32, M x K), source row-major with leading dim `ld` (elements).
// Layout: lanes 0-15 hold row m=lane, K {0..15};
//         lanes 16-31 hold row m=lane-16, K {16..31}.
__device__ __forceinline__ v16bf load_A16(const bf16* base, int ld, int lane) {
    int row = lane & 15;
    const bf16* p = base + (size_t)row * ld + ((lane & 16) ? 16 : 0);
    v8bf lo = *(const v8bf*)(p);
    v8bf hi = *(const v8bf*)(p + 8);
    v16bf r;
#pragma unroll
    for (int i = 0; i < 8; ++i) { r[i] = lo[i]; r[i + 8] = hi[i]; }
    return r;
}

// B operand (32x16, K x N), source laid out so B[k][n] = src_row(n)[k]:
// lane reads 16 contiguous halfs. `p` must already point at
// src + row(n=lane&15)*ld + k0 + (lane<16 ? 0 : 16).
__device__ __forceinline__ v16bf load_B16(const bf16* p) {
    v8bf lo = *(const v8bf*)(p);
    v8bf hi = *(const v8bf*)(p + 8);
    v16bf r;
#pragma unroll
    for (int i = 0; i < 8; ++i) { r[i] = lo[i]; r[i + 8] = hi[i]; }
    return r;
}

// ---------------------------------------------------------------------------
// Conversion kernels
// ---------------------------------------------------------------------------
__global__ void wha_cvt_bf16(const float* __restrict__ src, bf16* __restrict__ dst, int n) {
    int i = blockIdx.x * blockDim.x + threadIdx.x;
    if (i < n) dst[i] = (bf16)src[i];
}

// W [K][N] fp32  ->  Wt [N][K] bf16   (so B-operand reads are contiguous in k)
__global__ void wha_transpose_bf16(const float* __restrict__ W, bf16* __restrict__ Wt,
                                   int K, int N) {
    int idx = blockIdx.x * blockDim.x + threadIdx.x;
    if (idx >= K * N) return;
    int k = idx / N, n = idx % N;
    Wt[(size_t)n * K + k] = (bf16)W[idx];
}

// ---------------------------------------------------------------------------
// Fused QKV projection: xb[4096x1024] @ {WqT,WkT,WvT} + bias.
// Q,K -> bf16 row-major [4096][768]; V -> bf16 transposed Vt[b][h][64][S].
// Block = 128 threads = 4 waves; each wave computes a 32x64 tile. The k-loop
// is a 2x-unrolled ping-pong: no buffer-rotation moves, loads of one k-step
// always in flight under the WMMAs of the other.
// ---------------------------------------------------------------------------
__global__ void __launch_bounds__(128, 1)
wha_gemm_qkv(const bf16* __restrict__ xb,
             const bf16* __restrict__ WqT, const bf16* __restrict__ WkT,
             const bf16* __restrict__ WvT,
             const float* __restrict__ bq, const float* __restrict__ bk,
             const float* __restrict__ bv,
             bf16* __restrict__ Qb, bf16* __restrict__ Kb,
             bf16* __restrict__ Vt) {
    int lane = threadIdx.x & 31;
    int wave = threadIdx.x >> 5;
    int row0 = (blockIdx.x * 4 + wave) * 32;       // 0..4064, grid.x = 32
    int c0   = blockIdx.y * 64;                    // 0..2559

    const bf16* Wt; const float* bias; int n_base; int kind;
    if (c0 < STD_DIM)            { Wt = WqT; bias = bq; n_base = c0;               kind = 0; }
    else if (c0 < 2 * STD_DIM)   { Wt = WkT; bias = bk; n_base = c0 - STD_DIM;     kind = 1; }
    else                         { Wt = WvT; bias = bv; n_base = c0 - 2 * STD_DIM; kind = 2; }

    int kofs = (lane & 16) ? 16 : 0;
    int nrow = lane & 15;

    const bf16* ap0 = xb + (size_t)row0 * D_MODEL;
    const bf16* ap1 = ap0 + (size_t)16 * D_MODEL;
    const bf16* bptr[4];
#pragma unroll
    for (int t = 0; t < 4; ++t)
        bptr[t] = Wt + (size_t)(n_base + t * 16 + nrow) * D_MODEL + kofs;

    v8f acc[8];
#pragma unroll
    for (int t = 0; t < 8; ++t) acc[t] = zero8();

    v16bf aA0, aA1, bA[4];   // ping buffer
    v16bf aB0, aB1, bB[4];   // pong buffer

    auto load_step = [&](int k, v16bf& A0, v16bf& A1, v16bf* BB) {
        A0 = load_A16(ap0 + k, D_MODEL, lane);
        A1 = load_A16(ap1 + k, D_MODEL, lane);
#pragma unroll
        for (int t = 0; t < 4; ++t) BB[t] = load_B16(bptr[t] + k);
    };
    auto compute = [&](const v16bf& A0, const v16bf& A1, const v16bf* BB) {
#pragma unroll
        for (int t = 0; t < 4; ++t) {
            acc[t]     = wmma_bf16(A0, BB[t], acc[t]);
            acc[4 + t] = wmma_bf16(A1, BB[t], acc[4 + t]);
        }
    };

    load_step(0, aA0, aA1, bA);                       // step 0 -> ping
    for (int k0 = 0; k0 < D_MODEL - 64; k0 += 64) {
        load_step(k0 + 32, aB0, aB1, bB);             // step k0+32 -> pong
        compute(aA0, aA1, bA);                        // consume step k0
        load_step(k0 + 64, aA0, aA1, bA);             // step k0+64 -> ping
        compute(aB0, aB1, bB);                        // consume step k0+32
    }
    load_step(D_MODEL - 32, aB0, aB1, bB);            // step 992 -> pong
    compute(aA0, aA1, bA);                            // step 960
    compute(aB0, aB1, bB);                            // step 992

    int rofs = (lane & 16) ? 8 : 0;
    int coln = lane & 15;
#pragma unroll
    for (int rr = 0; rr < 2; ++rr) {
#pragma unroll
        for (int t = 0; t < 4; ++t) {
            int n = n_base + t * 16 + coln;
            float bb = bias[n];
#pragma unroll
            for (int r = 0; r < 8; ++r) {
                int row = row0 + rr * 16 + r + rofs;
                float val = acc[rr * 4 + t][r] + bb;
                if (kind == 2) {
                    int bidx = row >> 10, s = row & (SEQ - 1);
                    int hh = n >> 6, d = n & 63;
                    Vt[(((size_t)bidx * N_TOTAL + hh) * DK + d) * SEQ + s] = (bf16)val;
                } else {
                    bf16* dst = (kind == 0) ? Qb : Kb;
                    dst[(size_t)row * STD_DIM + n] = (bf16)val;
                }
            }
        }
    }
}

// ---------------------------------------------------------------------------
// Flash attention: one wave per (b, h, 16-query-row block).
// Std heads: scores via WMMA on Q/K bf16. Wave heads: VALU dot (chunk=6).
// Online softmax (f32), P*V via WMMA against transposed Vt. Next K/V block
// is prefetched (global_prefetch_b8) while softmax VALU runs.
// ---------------------------------------------------------------------------
__global__ void __launch_bounds__(32, 1)
wha_flash(const bf16* __restrict__ Qb, const bf16* __restrict__ Kb,
          const bf16* __restrict__ Vt,
          const float* __restrict__ wave_amp,
          bf16* __restrict__ attn) {
    int lane = threadIdx.x;
    int rb = blockIdx.x;          // 0..63
    int h  = blockIdx.y;          // 0..15
    int b  = blockIdx.z;          // 0..3
    int i0 = rb * 16;
    bool is_wave = (h < N_WAVE);

    __shared__ __align__(16) bf16 Pbuf[16 * 32];

    int rofs = (lane & 16) ? 8 : 0;
    int coln = lane & 15;
    int kofs = (lane & 16) ? 16 : 0;

    // Wave-head query rows cached per lane (its 8 rows of the C/D layout).
    float wi[8][WAVE_CHUNK];
    if (is_wave) {
#pragma unroll
        for (int r = 0; r < 8; ++r) {
            const float* wp = wave_amp + ((size_t)(b * SEQ + i0 + r + rofs) * WAVE_DIM) + h * WAVE_CHUNK;
#pragma unroll
            for (int c = 0; c < WAVE_CHUNK; ++c) wi[r][c] = wp[c];
        }
    }

    // Std-head Q operand is invariant over the key loop: load once.
    v16bf qa0, qa1;
    const bf16* Qbase = Qb + (size_t)(b * SEQ + i0) * STD_DIM + (h - N_WAVE) * DK;
    const bf16* Krow  = Kb + (size_t)(b * SEQ) * STD_DIM + (h - N_WAVE) * DK
                           + (size_t)coln * STD_DIM + kofs;
    if (!is_wave) {
        qa0 = load_A16(Qbase + 0,  STD_DIM, lane);
        qa1 = load_A16(Qbase + 32, STD_DIM, lane);
    }

    float mrow[8], lrow[8];
    v8f o[4];
#pragma unroll
    for (int r = 0; r < 8; ++r) { mrow[r] = -3.0e38f; lrow[r] = 0.0f; }
#pragma unroll
    for (int t = 0; t < 4; ++t) o[t] = zero8();

    const float scale_std  = 0.125f;          // 1/sqrt(64)
    const float scale_wave = 0.40824829046f;  // 1/sqrt(6)
    const bf16* Vbase = Vt + ((size_t)b * N_TOTAL + h) * DK * SEQ;
    const bf16* Vrow  = Vbase + (size_t)coln * SEQ + kofs;

    int jend = i0 + 16;                       // causal: keys j <= i0+15
    for (int j0 = 0; j0 < jend; j0 += 32) {
        // Prefetch next key/value block into cache while we compute this one.
        if (j0 + 32 < jend) {
            if (!is_wave) {
                const bf16* kn = Krow + (size_t)(j0 + 32) * STD_DIM;
                __builtin_prefetch(kn, 0, 3);
                __builtin_prefetch(kn + (size_t)16 * STD_DIM, 0, 3);
            }
            __builtin_prefetch(Vrow + j0 + 32, 0, 3);
            __builtin_prefetch(Vrow + (size_t)32 * SEQ + j0 + 32, 0, 3);
        }

        v8f s0 = zero8(), s1 = zero8();

        if (!is_wave) {
            const bf16* kp0 = Krow + (size_t)j0 * STD_DIM;
            const bf16* kp1 = kp0 + (size_t)16 * STD_DIM;
            s0 = wmma_bf16(qa0, load_B16(kp0),      s0);
            s1 = wmma_bf16(qa0, load_B16(kp1),      s1);
            s0 = wmma_bf16(qa1, load_B16(kp0 + 32), s0);
            s1 = wmma_bf16(qa1, load_B16(kp1 + 32), s1);
#pragma unroll
            for (int r = 0; r < 8; ++r) { s0[r] *= scale_std; s1[r] *= scale_std; }
        } else {
            float wj0[WAVE_CHUNK], wj1[WAVE_CHUNK];
            const float* wp0 = wave_amp + ((size_t)(b * SEQ + j0 + coln) * WAVE_DIM) + h * WAVE_CHUNK;
            const float* wp1 = wp0 + 16 * WAVE_DIM;
#pragma unroll
            for (int c = 0; c < WAVE_CHUNK; ++c) { wj0[c] = wp0[c]; wj1[c] = wp1[c]; }
#pragma unroll
            for (int r = 0; r < 8; ++r) {
                float d0 = 0.0f, d1 = 0.0f;
#pragma unroll
                for (int c = 0; c < WAVE_CHUNK; ++c) {
                    d0 = fmaf(wi[r][c], wj0[c], d0);
                    d1 = fmaf(wi[r][c], wj1[c], d1);
                }
                s0[r] = d0 * scale_wave;
                s1[r] = d1 * scale_wave;
            }
        }

        // Causal mask + online softmax update (row = r+rofs across 16 lanes).
#pragma unroll
        for (int r = 0; r < 8; ++r) {
            int qi = i0 + r + rofs;
            if (j0 + coln      > qi) s0[r] = -3.0e38f;
            if (j0 + 16 + coln > qi) s1[r] = -3.0e38f;

            float mx = fmaxf(s0[r], s1[r]);
#pragma unroll
            for (int m = 1; m < 16; m <<= 1) mx = fmaxf(mx, __shfl_xor(mx, m));
            float mnew = fmaxf(mrow[r], mx);
            float corr = __expf(mrow[r] - mnew);
            float p0 = __expf(s0[r] - mnew);
            float p1 = __expf(s1[r] - mnew);
            float sum = p0 + p1;
#pragma unroll
            for (int m = 1; m < 16; m <<= 1) sum += __shfl_xor(sum, m);
            lrow[r] = lrow[r] * corr + sum;
            mrow[r] = mnew;
#pragma unroll
            for (int t = 0; t < 4; ++t) o[t][r] *= corr;

            Pbuf[(r + rofs) * 32 + coln]      = (bf16)p0;
            Pbuf[(r + rofs) * 32 + 16 + coln] = (bf16)p1;
        }
        __syncthreads();   // single wave: orders LDS stores before A-layout reads

        v16bf pa = load_A16(Pbuf, 32, lane);
#pragma unroll
        for (int t = 0; t < 4; ++t) {
            const bf16* vp = Vrow + (size_t)(t * 16) * SEQ + j0;
            o[t] = wmma_bf16(pa, load_B16(vp), o[t]);
        }
        __syncthreads();   // protect Pbuf before next iteration overwrites it
    }

#pragma unroll
    for (int t = 0; t < 4; ++t) {
#pragma unroll
        for (int r = 0; r < 8; ++r) {
            float val = o[t][r] / lrow[r];
            int row = b * SEQ + i0 + r + rofs;
            attn[(size_t)row * D_MODEL + h * DK + t * 16 + coln] = (bf16)val;
        }
    }
}

// ---------------------------------------------------------------------------
// Output projection: attn[4096x1024] @ WoT + bo -> f32 out.
// Same 32x64-per-wave ping-pong scheme as the QKV GEMM.
// ---------------------------------------------------------------------------
__global__ void __launch_bounds__(128, 1)
wha_gemm_out(const bf16* __restrict__ attn, const bf16* __restrict__ WoT,
             const float* __restrict__ bo, float* __restrict__ out) {
    int lane = threadIdx.x & 31;
    int wave = threadIdx.x >> 5;
    int row0 = (blockIdx.x * 4 + wave) * 32;   // grid.x = 32
    int n0   = blockIdx.y * 64;

    int kofs = (lane & 16) ? 16 : 0;
    int nrow = lane & 15;

    const bf16* ap0 = attn + (size_t)row0 * D_MODEL;
    const bf16* ap1 = ap0 + (size_t)16 * D_MODEL;
    const bf16* bptr[4];
#pragma unroll
    for (int t = 0; t < 4; ++t)
        bptr[t] = WoT + (size_t)(n0 + t * 16 + nrow) * D_MODEL + kofs;

    v8f acc[8];
#pragma unroll
    for (int t = 0; t < 8; ++t) acc[t] = zero8();

    v16bf aA0, aA1, bA[4];
    v16bf aB0, aB1, bB[4];

    auto load_step = [&](int k, v16bf& A0, v16bf& A1, v16bf* BB) {
        A0 = load_A16(ap0 + k, D_MODEL, lane);
        A1 = load_A16(ap1 + k, D_MODEL, lane);
#pragma unroll
        for (int t = 0; t < 4; ++t) BB[t] = load_B16(bptr[t] + k);
    };
    auto compute = [&](const v16bf& A0, const v16bf& A1, const v16bf* BB) {
#pragma unroll
        for (int t = 0; t < 4; ++t) {
            acc[t]     = wmma_bf16(A0, BB[t], acc[t]);
            acc[4 + t] = wmma_bf16(A1, BB[t], acc[4 + t]);
        }
    };

    load_step(0, aA0, aA1, bA);
    for (int k0 = 0; k0 < D_MODEL - 64; k0 += 64) {
        load_step(k0 + 32, aB0, aB1, bB);
        compute(aA0, aA1, bA);
        load_step(k0 + 64, aA0, aA1, bA);
        compute(aB0, aB1, bB);
    }
    load_step(D_MODEL - 32, aB0, aB1, bB);
    compute(aA0, aA1, bA);
    compute(aB0, aB1, bB);

    int rofs = (lane & 16) ? 8 : 0;
    int coln = lane & 15;
#pragma unroll
    for (int rr = 0; rr < 2; ++rr) {
#pragma unroll
        for (int t = 0; t < 4; ++t) {
            int n = n0 + t * 16 + coln;
            float bb = bo[n];
#pragma unroll
            for (int r = 0; r < 8; ++r) {
                out[(size_t)(row0 + rr * 16 + r + rofs) * D_MODEL + n] = acc[rr * 4 + t][r] + bb;
            }
        }
    }
}

// ---------------------------------------------------------------------------
// Host-side launcher
// ---------------------------------------------------------------------------
extern "C" void kernel_launch(void* const* d_in, const int* in_sizes, int n_in,
                              void* d_out, int out_size, void* d_ws, size_t ws_size,
                              hipStream_t stream) {
    const float* x        = (const float*)d_in[0];
    const float* wave_amp = (const float*)d_in[1];
    // d_in[2] = causal_mask (computed arithmetically; unused)
    const float* Wq = (const float*)d_in[3];
    const float* bq = (const float*)d_in[4];
    const float* Wk = (const float*)d_in[5];
    const float* bk = (const float*)d_in[6];
    const float* Wv = (const float*)d_in[7];
    const float* bv = (const float*)d_in[8];
    const float* Wo = (const float*)d_in[9];
    const float* bo = (const float*)d_in[10];
    float* out = (float*)d_out;

    char* ws = (char*)d_ws;
    bf16* xb   = (bf16*)(ws);             //  8,388,608 B : x bf16 [4096][1024]
    bf16* Qb   = (bf16*)(ws + 8388608);   //  6,291,456 B : Q bf16 [4096][768]
    bf16* Kb   = (bf16*)(ws + 14680064);  //  6,291,456 B : K bf16 [4096][768]
    bf16* VtB  = (bf16*)(ws + 20971520);  //  8,388,608 B : V^T bf16 [4][16][64][1024]
    bf16* attn = (bf16*)(ws + 29360128);  //  8,388,608 B : attn bf16 [4096][1024]
    bf16* WqT  = (bf16*)(ws + 37748736);  //  1,572,864 B : Wq^T bf16 [768][1024]
    bf16* WkT  = (bf16*)(ws + 39321600);  //  1,572,864 B
    bf16* WvT  = (bf16*)(ws + 40894464);  //  2,097,152 B : Wv^T bf16 [1024][1024]
    bf16* WoT  = (bf16*)(ws + 42991616);  //  2,097,152 B  (total 45,088,768 B)

    int nx = BATCH * SEQ * D_MODEL;
    wha_cvt_bf16<<<(nx + 255) / 256, 256, 0, stream>>>(x, xb, nx);
    wha_transpose_bf16<<<(D_MODEL * STD_DIM + 255) / 256, 256, 0, stream>>>(Wq, WqT, D_MODEL, STD_DIM);
    wha_transpose_bf16<<<(D_MODEL * STD_DIM + 255) / 256, 256, 0, stream>>>(Wk, WkT, D_MODEL, STD_DIM);
    wha_transpose_bf16<<<(D_MODEL * D_MODEL + 255) / 256, 256, 0, stream>>>(Wv, WvT, D_MODEL, D_MODEL);
    wha_transpose_bf16<<<(D_MODEL * D_MODEL + 255) / 256, 256, 0, stream>>>(Wo, WoT, D_MODEL, D_MODEL);

    wha_gemm_qkv<<<dim3(32, 40), 128, 0, stream>>>(xb, WqT, WkT, WvT, bq, bk, bv, Qb, Kb, VtB);
    wha_flash<<<dim3(64, 16, 4), 32, 0, stream>>>(Qb, Kb, VtB, wave_amp, attn);
    wha_gemm_out<<<dim3(32, 16), 128, 0, stream>>>(attn, WoT, bo, out);
}
